// CNN_LSTM_METRICS_34385508171791
// MI455X (gfx1250) — matrix operand order repaired and checked by reading, not compiled, not run on hardware
//
#include <hip/hip_runtime.h>
#include <hip/hip_bf16.h>
#include <math.h>

// ---------------- problem constants (match reference) ----------------
constexpr int CB   = 64;      // batch
constexpr int CL   = 3000;    // text length
constexpr int CK   = 5;       // conv kernel / pool
constexpr int CC1  = 16;
constexpr int CC2  = 32;
constexpr int CT1  = CL - CK + 1;        // 2996
constexpr int CP1  = CT1 / CK;           // 599
constexpr int CT2  = CP1 - CK + 1;       // 595
constexpr int CP2  = CT2 / CK;           // 119  (= F_LSTM)
constexpr int CH   = 100;     // LSTM hidden
constexpr int CSEQ = CC2;     // 32 LSTM steps
constexpr int CM   = 20;
constexpr int CD   = 128;     // node feature dim
constexpr int CN   = 100000;  // nodes
constexpr int CE   = 600000;  // edges
constexpr int CR   = 18;      // relations
constexpr int CNL  = 5;       // rgcn layers
constexpr int MW   = 4;       // waves per WMMA block (128 threads)

typedef __bf16 bhalf;
typedef __attribute__((ext_vector_type(16))) __bf16 v16bf;
typedef __attribute__((ext_vector_type(8)))  float  v8f;
typedef __attribute__((ext_vector_type(4)))  int    v4i;
typedef unsigned int u32;

#define AS1 __attribute__((address_space(1)))
#define AS3 __attribute__((address_space(3)))

#if defined(__HIP_DEVICE_COMPILE__) && defined(__has_builtin)
#if __has_builtin(__builtin_amdgcn_global_load_async_to_lds_b128)
#define HAVE_ASYNC_LDS 1
#endif
#endif
#ifndef HAVE_ASYNC_LDS
#define HAVE_ASYNC_LDS 0
#endif

// ======================= CNN path =======================
__global__ void k_conv1(const float* __restrict__ x, const float* __restrict__ w,
                        const float* __restrict__ bias, float* __restrict__ y) {
  int i = blockIdx.x * 256 + threadIdx.x;
  if (i >= CB * CC1 * CT1) return;
  int t = i % CT1, c = (i / CT1) % CC1, b = i / (CT1 * CC1);
  const float* xp = x + b * CL + t;
  float s = bias[c];
#pragma unroll
  for (int k = 0; k < CK; k++) s += xp[k] * w[c * CK + k];
  y[i] = s;
}

__global__ void k_bnstats(const float* __restrict__ x, float* __restrict__ stat,
                          int C, int T) {
  int c = blockIdx.x;
  __shared__ float ssum[256], ssq[256];
  float s = 0.f, q = 0.f;
  int len = CB * T;
  for (int i = threadIdx.x; i < len; i += 256) {
    int b = i / T, t = i % T;
    float v = x[(b * C + c) * T + t];
    s += v; q += v * v;
  }
  ssum[threadIdx.x] = s; ssq[threadIdx.x] = q;
  __syncthreads();
  for (int o = 128; o > 0; o >>= 1) {
    if (threadIdx.x < o) {
      ssum[threadIdx.x] += ssum[threadIdx.x + o];
      ssq[threadIdx.x]  += ssq[threadIdx.x + o];
    }
    __syncthreads();
  }
  if (threadIdx.x == 0) {
    float mean = ssum[0] / (float)len;
    float var  = ssq[0] / (float)len - mean * mean;
    stat[c * 2] = mean;
    stat[c * 2 + 1] = rsqrtf(var + 1e-7f);
  }
}

__global__ void k_bnpool(const float* __restrict__ x, const float* __restrict__ stat,
                         const float* __restrict__ g, const float* __restrict__ be,
                         float* __restrict__ y, int C, int T, int TP) {
  int i = blockIdx.x * 256 + threadIdx.x;
  if (i >= CB * C * TP) return;
  int tp = i % TP, c = (i / TP) % C, b = i / (TP * C);
  float mean = stat[c * 2], r = stat[c * 2 + 1], gg = g[c], bb = be[c];
  float m = 0.f;
#pragma unroll
  for (int k = 0; k < CK; k++) {
    float v = (x[(b * C + c) * T + tp * CK + k] - mean) * r * gg + bb;
    m = fmaxf(m, fmaxf(v, 0.f));
  }
  y[i] = m;
}

__global__ void k_conv2(const float* __restrict__ x, const float* __restrict__ w,
                        const float* __restrict__ bias, float* __restrict__ y) {
  int i = blockIdx.x * 256 + threadIdx.x;
  if (i >= CB * CC2 * CT2) return;
  int t = i % CT2, c = (i / CT2) % CC2, b = i / (CT2 * CC2);
  float s = bias[c];
  for (int ic = 0; ic < CC1; ic++) {
    const float* xp = x + (b * CC1 + ic) * CP1 + t;
    const float* wp = w + (c * CC1 + ic) * CK;
#pragma unroll
    for (int k = 0; k < CK; k++) s += xp[k] * wp[k];
  }
  y[i] = s;
}

// ======================= LSTM =======================
__global__ void k_xw(const float* __restrict__ pool2, const float* __restrict__ Wih,
                     float* __restrict__ XW) {
  int i = blockIdx.x * 256 + threadIdx.x;
  if (i >= CSEQ * CB * 4 * CH) return;
  int u = i % (4 * CH), b = (i / (4 * CH)) % CB, t = i / (4 * CH * CB);
  const float* xp = pool2 + (b * CC2 + t) * CP2;
  const float* wp = Wih + u * CP2;
  float s = 0.f;
  for (int f = 0; f < CP2; f++) s += xp[f] * wp[f];
  XW[i] = s;
}

__global__ void k_lstm(const float* __restrict__ XW, const float* __restrict__ Whh,
                       const float* __restrict__ bih, const float* __restrict__ bhh,
                       float* __restrict__ hidden) {
  __shared__ float hbuf[2][CB * CH];   // 51.2 KB
  const int tid = threadIdx.x;         // 640 threads, 10 pairs each
  float creg[10];
#pragma unroll
  for (int p = 0; p < 10; p++) { creg[p] = 0.f; hbuf[0][tid + p * 640] = 0.f; }
  __syncthreads();
  for (int t = 0; t < CSEQ; t++) {
    const float* hp = hbuf[t & 1];
    float* hn = hbuf[(t + 1) & 1];
    const float* xw = XW + (size_t)t * CB * 4 * CH;
#pragma unroll
    for (int p = 0; p < 10; p++) {
      int q = tid + p * 640;
      int b = q / CH, u = q % CH;
      float g4[4];
#pragma unroll
      for (int j = 0; j < 4; j++) {
        int row = j * CH + u;
        float s = xw[b * 4 * CH + row] + bih[row] + bhh[row];
        const float* wr = Whh + row * CH;
        const float* hb = hp + b * CH;
        for (int k = 0; k < CH; k++) s += hb[k] * wr[k];
        g4[j] = s;
      }
      float ig = 1.f / (1.f + expf(-g4[0]));
      float fg = 1.f / (1.f + expf(-g4[1]));
      float gg = tanhf(g4[2]);
      float og = 1.f / (1.f + expf(-g4[3]));
      float c = fg * creg[p] + ig * gg;
      creg[p] = c;
      hn[b * CH + u] = og * tanhf(c);
    }
    __syncthreads();
  }
#pragma unroll
  for (int p = 0; p < 10; p++) { int q = tid + p * 640; hidden[q] = hbuf[0][q]; }
}

// ======================= metrics FC =======================
__global__ void k_metrics(const float* __restrict__ x, const float* __restrict__ w,
                          const float* __restrict__ b, float* __restrict__ y) {
  int i = blockIdx.x * 256 + threadIdx.x;
  if (i >= CB * 24) return;
  int j = i % 24, bb = i / 24;
  float s = b[j];
  for (int k = 0; k < CM; k++) s += x[bb * CM + k] * w[k * 24 + j];
  y[i] = fmaxf(s, 0.f);
}

// ======================= RGCN prep =======================
__global__ void k_gather(const float* __restrict__ emb, const int* __restrict__ xn,
                         float* __restrict__ hA, bhalf* __restrict__ hbf) {
  int i = blockIdx.x * 256 + threadIdx.x;
  if (i >= CN * CD) return;
  int n = i >> 7, d = i & 127;
  float v = emb[(size_t)xn[n] * CD + d];
  hA[i] = v;
  hbf[i] = (bhalf)v;
}

__global__ void k_cnt(const int* __restrict__ edst, const int* __restrict__ et,
                      int* __restrict__ cnt, int* __restrict__ bc) {
  int e = blockIdx.x * 256 + threadIdx.x;
  if (e >= CE) return;
  atomicAdd(&cnt[(size_t)edst[e] * CR + et[e]], 1);
  atomicAdd(&bc[et[e]], 1);
}

__global__ void k_invw(const int* __restrict__ edst, const int* __restrict__ et,
                       const int* __restrict__ cnt, float* __restrict__ invw) {
  int e = blockIdx.x * 256 + threadIdx.x;
  if (e >= CE) return;
  int c = cnt[(size_t)edst[e] * CR + et[e]];
  invw[e] = 1.0f / (float)(c > 0 ? c : 1);
}

__global__ void k_scan(const int* __restrict__ bc, int* __restrict__ eoff,
                       int* __restrict__ boff) {
  if (threadIdx.x == 0 && blockIdx.x == 0) {
    int e = 0, b = 0;
    eoff[0] = 0; boff[0] = 0;
    for (int r = 0; r < CR; r++) {
      e += bc[r]; eoff[r + 1] = e;
      int groups = (bc[r] + 15) >> 4;
      b += (groups + MW - 1) / MW;
      boff[r + 1] = b;
    }
  }
}

__global__ void k_scatter(const int* __restrict__ et, const int* __restrict__ eoff,
                          int* __restrict__ cursor, int* __restrict__ sidx) {
  int e = blockIdx.x * 256 + threadIdx.x;
  if (e >= CE) return;
  int r = et[e];
  int pos = eoff[r] + atomicAdd(&cursor[r], 1);
  sidx[pos] = e;
}

// f32 -> bf16 with transpose: dst[mat][n][k] = (bf16) src[mat][k][n]
__global__ void k_f2bf_t(const float* __restrict__ src, bhalf* __restrict__ dst,
                         int nmat) {
  int i = blockIdx.x * 256 + threadIdx.x;
  if (i >= nmat * CD * CD) return;
  int mat = i >> 14, rem = i & 16383;
  int k = rem >> 7, n = rem & 127;
  dst[((size_t)mat << 14) + n * CD + k] = (bhalf)src[i];
}

// ============== WMMA helpers (wave32, v_wmma_f32_16x16x32_bf16) ==============
// A tile 16x32 bf16 (contiguous k order): lane kh*16+m holds, per K-chunk kc, pairs at
// K = kc*32 + kh*16 + v*2 + sub.  We stage A pre-swizzled in
// LDS (u32 pair granularity) so each lane's per-chunk fragment is one
// contiguous 32B run -> ds_load_b128 x2.
__device__ inline int a_swz_index(int row, int kp /* pair 0..63 */) {
  int kc = kp >> 4, kpl = kp & 15;
  int kh = (kpl >> 3) & 1;
  int v  = kpl & 7;
  return kc * 256 + (kh * 16 + row) * 8 + v;   // u32 units
}

// B tile 32x16 bf16 from sW[n*128 + k] (transposed W): lane kh*16+(n&15),
// K = kc*32 + kh*16 + 2v+sub -> contiguous 32B per lane.
__device__ inline v16bf load_b_frag(const bhalf* sW, int lane, int nt, int kc) {
  int n = nt * 16 + (lane & 15), kh = (lane >> 4) & 1;
  return *(const v16bf*)(sW + n * CD + kc * 32 + kh * 16);
}

// Stage the (pre-transposed) 128x128 bf16 weight block into LDS.
__device__ inline void stage_w(const bhalf* __restrict__ Ws, bhalf* sW, int tid) {
#if HAVE_ASYNC_LDS
  // 32KB via async global->LDS b128, tracked by ASYNCcnt.
  const char* gp = (const char*)Ws;
  char* lp = (char*)sW;
#pragma unroll
  for (int it = 0; it < 16; ++it) {
    int byteoff = (tid + it * 128) * 16;
    __builtin_amdgcn_global_load_async_to_lds_b128(
        (AS1 v4i*)(gp + byteoff), (AS3 v4i*)(lp + byteoff), 0, 0);
  }
  asm volatile("s_wait_asynccnt 0x0" ::: "memory");
#else
  const uint4* gp = (const uint4*)Ws;
  uint4* lp = (uint4*)sW;
  for (int it = tid; it < 2048; it += 128) lp[it] = gp[it];
#endif
}

// ======================= RGCN root: out = h @ Wroot + bias =======================
__global__ __launch_bounds__(128) void k_root(const bhalf* __restrict__ hbf,
                                              const bhalf* __restrict__ Wsrc,
                                              const float* __restrict__ bias,
                                              float* __restrict__ out) {
  __shared__ __align__(16) bhalf sW[CD * CD];        // pre-transposed weights
  __shared__ __align__(16) u32   sA[MW][16 * 64];    // swizzled A pairs
  int tid = threadIdx.x;
  __builtin_prefetch(Wsrc + (size_t)tid * 128, 0, 0);
  stage_w(Wsrc, sW, tid);
  __syncthreads();
  int wave = tid >> 5, lane = tid & 31;
  int group = blockIdx.x * MW + wave;
  if (group >= CN / 16) return;
  u32* slab = sA[wave];
  const u32* hb = (const u32*)hbf;
  for (int u = lane; u < 16 * 64; u += 32) {
    int row = u >> 6, kp = u & 63;
    slab[a_swz_index(row, kp)] = hb[(size_t)(group * 16 + row) * 64 + kp];
  }
  int kh = (lane >> 4) & 1;
  v16bf a[4];
#pragma unroll
  for (int kc = 0; kc < 4; kc++)
    a[kc] = *(const v16bf*)(slab + kc * 256 + lane * 8);
#pragma unroll
  for (int nt = 0; nt < 8; nt++) {
    int n = nt * 16 + (lane & 15);
    v8f acc = {0.f, 0.f, 0.f, 0.f, 0.f, 0.f, 0.f, 0.f};
#pragma unroll
    for (int kc = 0; kc < 4; kc++) {
      v16bf b = load_b_frag(sW, lane, nt, kc);
      acc = __builtin_amdgcn_wmma_f32_16x16x32_bf16(false, a[kc], false, b,
                                                    (short)0, acc, false, false);
    }
    float bn = bias[n];
#pragma unroll
    for (int g = 0; g < 8; g++) {
      int row = group * 16 + g + kh * 8;
      out[(size_t)row * CD + n] = acc[g] + bn;
    }
  }
}

// ============ RGCN messages: out[dst] += (h[src] @ Wrel[etype]) * invw ============
__global__ __launch_bounds__(128) void k_msg(const bhalf* __restrict__ hbf,
                                             const bhalf* __restrict__ Wbf, // 18 mats, transposed
                                             const int* __restrict__ sidx,
                                             const int* __restrict__ esrc,
                                             const int* __restrict__ edst,
                                             const float* __restrict__ invw,
                                             const int* __restrict__ eoff,
                                             const int* __restrict__ boff,
                                             const int* __restrict__ bc,
                                             float* __restrict__ out) {
  __shared__ __align__(16) bhalf sW[CD * CD];
  __shared__ __align__(16) u32   sA[MW][16 * 64];
  int blk = blockIdx.x;
  if (blk >= boff[CR]) return;
  int r = 0;
  while (blk >= boff[r + 1]) r++;
  int jb = blk - boff[r];
  int tid = threadIdx.x;
  const bhalf* Ws = Wbf + ((size_t)r << 14);
  __builtin_prefetch(Ws + (size_t)tid * 128, 0, 0);
  stage_w(Ws, sW, tid);
  __syncthreads();
  int wave = tid >> 5, lane = tid & 31;
  int groups_r = (bc[r] + 15) >> 4;
  int j = jb * MW + wave;
  if (j >= groups_r) return;
  int ebase = eoff[r] + j * 16;
  int valid = bc[r] - j * 16; if (valid > 16) valid = 16;
  u32* slab = sA[wave];
  const u32* hb = (const u32*)hbf;
  for (int u = lane; u < 16 * 64; u += 32) {
    int row = u >> 6, kp = u & 63;
    u32 val = 0;
    if (row < valid) {
      int e = sidx[ebase + row];
      val = hb[(size_t)esrc[e] * 64 + kp];
    }
    slab[a_swz_index(row, kp)] = val;
  }
  int kh = (lane >> 4) & 1;
  int drow[8]; float wrow[8];
#pragma unroll
  for (int g = 0; g < 8; g++) {
    int mm = g + kh * 8;
    if (mm < valid) {
      int e = sidx[ebase + mm];
      drow[g] = edst[e];
      wrow[g] = invw[e];
    } else drow[g] = -1;
  }
  v16bf a[4];
#pragma unroll
  for (int kc = 0; kc < 4; kc++)
    a[kc] = *(const v16bf*)(slab + kc * 256 + lane * 8);
#pragma unroll
  for (int nt = 0; nt < 8; nt++) {
    int n = nt * 16 + (lane & 15);
    v8f acc = {0.f, 0.f, 0.f, 0.f, 0.f, 0.f, 0.f, 0.f};
#pragma unroll
    for (int kc = 0; kc < 4; kc++) {
      v16bf b = load_b_frag(sW, lane, nt, kc);
      acc = __builtin_amdgcn_wmma_f32_16x16x32_bf16(false, a[kc], false, b,
                                                    (short)0, acc, false, false);
    }
#pragma unroll
    for (int g = 0; g < 8; g++) {
      if (drow[g] >= 0)
        atomicAdd(&out[(size_t)drow[g] * CD + n], acc[g] * wrow[g]);
    }
  }
}

__global__ void k_relu_conv(const float* __restrict__ x, float* __restrict__ hA,
                            bhalf* __restrict__ hbf) {
  int i = blockIdx.x * 256 + threadIdx.x;
  if (i >= CN * CD) return;
  float v = fmaxf(x[i], 0.f);
  hA[i] = v;
  hbf[i] = (bhalf)v;
}

// ========== graph mean pool (batch sorted -> binary search, deterministic) ==========
__global__ void k_pool(const float* __restrict__ h, const int* __restrict__ batch,
                       float* __restrict__ graph) {
  int b = blockIdx.x;       // 64 blocks
  int d = threadIdx.x;      // 128 threads
  int lo, hi;
  { int l = 0, r = CN; while (l < r) { int m = (l + r) >> 1; if (batch[m] < b) l = m + 1; else r = m; } lo = l; }
  { int l = 0, r = CN; while (l < r) { int m = (l + r) >> 1; if (batch[m] < b + 1) l = m + 1; else r = m; } hi = l; }
  float s = 0.f;
  for (int row = lo; row < hi; row++) s += h[(size_t)row * CD + d];
  int c = hi - lo;
  graph[b * CD + d] = s / (float)(c > 0 ? c : 1);
}

// ======================= head MLP =======================
__global__ void k_head(const float* __restrict__ hidden, const float* __restrict__ met,
                       const float* __restrict__ graph,
                       const float* __restrict__ w1, const float* __restrict__ b1,
                       const float* __restrict__ w2, const float* __restrict__ b2,
                       const float* __restrict__ w3, const float* __restrict__ b3,
                       float* __restrict__ outp) {
  __shared__ float s1[CB * 32];
  __shared__ float s2[CB * 16];
  int tid = threadIdx.x;  // 256
  for (int i = tid; i < CB * 32; i += 256) {
    int b = i >> 5, j = i & 31;
    float s = b1[j];
    for (int k = 0; k < CH; k++)  s += hidden[b * CH + k] * w1[k * 32 + j];
    for (int k = 0; k < 24; k++)  s += met[b * 24 + k]    * w1[(CH + k) * 32 + j];
    for (int k = 0; k < CD; k++)  s += graph[b * CD + k]  * w1[(CH + 24 + k) * 32 + j];
    s1[i] = fmaxf(s, 0.f);
  }
  __syncthreads();
  for (int i = tid; i < CB * 16; i += 256) {
    int b = i >> 4, j = i & 15;
    float s = b2[j];
    for (int k = 0; k < 32; k++) s += s1[b * 32 + k] * w2[k * 16 + j];
    s2[i] = fmaxf(s, 0.f);
  }
  __syncthreads();
  for (int i = tid; i < CB; i += 256) {
    float s = b3[0];
    for (int k = 0; k < 16; k++) s += s2[i * 16 + k] * w3[k];
    outp[i] = s;
  }
}

// ======================= host launcher =======================
extern "C" void kernel_launch(void* const* d_in, const int* in_sizes, int n_in,
                              void* d_out, int out_size, void* d_ws, size_t ws_size,
                              hipStream_t stream) {
  const float* text    = (const float*)d_in[0];
  const float* metrics = (const float*)d_in[1];
  const int*   x_nodes = (const int*)d_in[2];
  const int*   eidx    = (const int*)d_in[3];
  const int*   etype   = (const int*)d_in[4];
  const int*   batch   = (const int*)d_in[5];
  const float* emb     = (const float*)d_in[6];
  const float* conv1_w = (const float*)d_in[7];
  const float* conv1_b = (const float*)d_in[8];
  const float* bn1_g   = (const float*)d_in[9];
  const float* bn1_b   = (const float*)d_in[10];
  const float* conv2_w = (const float*)d_in[11];
  const float* conv2_b = (const float*)d_in[12];
  const float* bn2_g   = (const float*)d_in[13];
  const float* bn2_b   = (const float*)d_in[14];
  const float* lstm_Wih = (const float*)d_in[15];
  const float* lstm_Whh = (const float*)d_in[16];
  const float* lstm_bih = (const float*)d_in[17];
  const float* lstm_bhh = (const float*)d_in[18];
  const float* mfc_w   = (const float*)d_in[19];
  const float* mfc_b   = (const float*)d_in[20];
  const float* rgcn_W  = (const float*)d_in[21];
  const float* rgcn_rt = (const float*)d_in[22];
  const float* rgcn_b  = (const float*)d_in[23];
  const float* fc1_w   = (const float*)d_in[24];
  const float* fc1_b   = (const float*)d_in[25];
  const float* fc2_w   = (const float*)d_in[26];
  const float* fc2_b   = (const float*)d_in[27];
  const float* fc3_w   = (const float*)d_in[28];
  const float* fc3_b   = (const float*)d_in[29];

  const int* esrc = eidx;
  const int* edst = eidx + CE;

  char* base = (char*)d_ws;
  size_t off = 0;
  auto alloc = [&](size_t bytes) -> void* {
    void* p = (void*)(base + off);
    off += (bytes + 255) & ~(size_t)255;
    return p;
  };
  float* conv1o = (float*)alloc((size_t)CB * CC1 * CT1 * 4);
  float* stat1  = (float*)alloc(CC1 * 2 * 4);
  float* pool1  = (float*)alloc((size_t)CB * CC1 * CP1 * 4);
  float* conv2o = (float*)alloc((size_t)CB * CC2 * CT2 * 4);
  float* stat2  = (float*)alloc(CC2 * 2 * 4);
  float* pool2  = (float*)alloc((size_t)CB * CC2 * CP2 * 4);
  float* XW     = (float*)alloc((size_t)CSEQ * CB * 4 * CH * 4);
  float* hidden = (float*)alloc((size_t)CB * CH * 4);
  float* met    = (float*)alloc((size_t)CB * 24 * 4);
  float* hA     = (float*)alloc((size_t)CN * CD * 4);
  float* hB     = (float*)alloc((size_t)CN * CD * 4);
  bhalf* hbf    = (bhalf*)alloc((size_t)CN * CD * 2);
  bhalf* Wrelbf = (bhalf*)alloc((size_t)CNL * CR * CD * CD * 2);
  bhalf* rootbf = (bhalf*)alloc((size_t)CNL * CD * CD * 2);
  int*   cnt    = (int*)alloc((size_t)CN * CR * 4);
  float* invw   = (float*)alloc((size_t)CE * 4);
  int*   bc     = (int*)alloc(CR * 4);
  int*   eoff   = (int*)alloc((CR + 1) * 4);
  int*   boff   = (int*)alloc((CR + 1) * 4);
  int*   cursor = (int*)alloc(CR * 4);
  int*   sidx   = (int*)alloc((size_t)CE * 4);
  float* graph  = (float*)alloc((size_t)CB * CD * 4);
  (void)off; (void)ws_size; (void)n_in; (void)in_sizes; (void)out_size;

  auto cdiv = [](long long a, long long b) { return (int)((a + b - 1) / b); };

  // ---- CNN ----
  k_conv1<<<cdiv((long long)CB * CC1 * CT1, 256), 256, 0, stream>>>(text, conv1_w, conv1_b, conv1o);
  k_bnstats<<<CC1, 256, 0, stream>>>(conv1o, stat1, CC1, CT1);
  k_bnpool<<<cdiv((long long)CB * CC1 * CP1, 256), 256, 0, stream>>>(conv1o, stat1, bn1_g, bn1_b, pool1, CC1, CT1, CP1);
  k_conv2<<<cdiv((long long)CB * CC2 * CT2, 256), 256, 0, stream>>>(pool1, conv2_w, conv2_b, conv2o);
  k_bnstats<<<CC2, 256, 0, stream>>>(conv2o, stat2, CC2, CT2);
  k_bnpool<<<cdiv((long long)CB * CC2 * CP2, 256), 256, 0, stream>>>(conv2o, stat2, bn2_g, bn2_b, pool2, CC2, CT2, CP2);

  // ---- LSTM ----
  k_xw<<<cdiv((long long)CSEQ * CB * 4 * CH, 256), 256, 0, stream>>>(pool2, lstm_Wih, XW);
  k_lstm<<<1, 640, 0, stream>>>(XW, lstm_Whh, lstm_bih, lstm_bhh, hidden);

  // ---- metrics ----
  k_metrics<<<cdiv((long long)CB * 24, 256), 256, 0, stream>>>(metrics, mfc_w, mfc_b, met);

  // ---- RGCN prep ----
  k_gather<<<cdiv((long long)CN * CD, 256), 256, 0, stream>>>(emb, x_nodes, hA, hbf);
  (void)hipMemsetAsync(cnt, 0, (size_t)CN * CR * 4, stream);
  (void)hipMemsetAsync(bc, 0, CR * 4, stream);
  (void)hipMemsetAsync(cursor, 0, CR * 4, stream);
  k_cnt<<<cdiv(CE, 256), 256, 0, stream>>>(edst, etype, cnt, bc);
  k_invw<<<cdiv(CE, 256), 256, 0, stream>>>(edst, etype, cnt, invw);
  k_scan<<<1, 1, 0, stream>>>(bc, eoff, boff);
  k_scatter<<<cdiv(CE, 256), 256, 0, stream>>>(etype, eoff, cursor, sidx);
  k_f2bf_t<<<cdiv((long long)CNL * CR * CD * CD, 256), 256, 0, stream>>>(rgcn_W, Wrelbf, CNL * CR);
  k_f2bf_t<<<cdiv((long long)CNL * CD * CD, 256), 256, 0, stream>>>(rgcn_rt, rootbf, CNL);

  // ---- RGCN layers ----
  const int nbRoot = cdiv(CN / 16, MW);
  const int nbMsg  = CE / (16 * MW) + 32;   // >= boff[CR] upper bound; extras exit early
  for (int l = 0; l < CNL; l++) {
    k_root<<<nbRoot, 128, 0, stream>>>(hbf, rootbf + ((size_t)l << 14),
                                       rgcn_b + (size_t)l * CD, hB);
    k_msg<<<nbMsg, 128, 0, stream>>>(hbf, Wrelbf + ((size_t)l * CR << 14),
                                     sidx, esrc, edst, invw, eoff, boff, bc, hB);
    k_relu_conv<<<cdiv((long long)CN * CD, 256), 256, 0, stream>>>(hB, hA, hbf);
  }

  // ---- pool + head ----
  k_pool<<<CB, CD, 0, stream>>>(hA, batch, graph);
  k_head<<<1, 256, 0, stream>>>(hidden, met, graph, fc1_w, fc1_b, fc2_w, fc2_b,
                                fc3_w, fc3_b, (float*)d_out);
}
